// MCInfoNCE_57054345560167
// MI455X (gfx1250) — matrix operand order repaired and checked by reading, not compiled, not run on hardware
//
#include <hip/hip_runtime.h>

#define BB 4096
#define DD 128
#define SS 8
#define JT 64                       // columns staged per LDS panel
#define PANEL_ELEMS (JT * DD)       // 8192 bf16 = 16 KB

typedef __attribute__((ext_vector_type(16))) __bf16 v16bf;
typedef __attribute__((ext_vector_type(8)))  __bf16 v8bf;
typedef __attribute__((ext_vector_type(8)))  float  v8f;

// ---------------- CDNA5 async copy helpers (inline asm, ISA 15.18.3 op 98) ----------------
__device__ __forceinline__ void async_g2l_b128(unsigned lds_off, const void* gaddr) {
    asm volatile("global_load_async_to_lds_b128 %0, %1, off"
                 :: "v"(lds_off), "v"(gaddr) : "memory");
}
#define S_WAIT_ASYNCCNT(n) asm volatile("s_wait_asynccnt %0" :: "i"(n) : "memory")

// ---------------- RNG helpers ----------------
__device__ __forceinline__ unsigned hashu(unsigned x) {
    x ^= x >> 17; x *= 0xed5ad4bbu;
    x ^= x >> 11; x *= 0xac4c1b51u;
    x ^= x >> 15; x *= 0x31848babu;
    x ^= x >> 14; return x;
}
__device__ __forceinline__ float u01(unsigned h) {
    return ((h >> 8) + 0.5f) * (1.0f / 16777216.0f);
}

__device__ __forceinline__ float blockReduceSum128(float v, float* red) {
    #pragma unroll
    for (int m = 16; m >= 1; m >>= 1) v += __shfl_xor(v, m, 32);
    const int w = threadIdx.x >> 5;
    __syncthreads();
    if ((threadIdx.x & 31) == 0) red[w] = v;
    __syncthreads();
    return red[0] + red[1] + red[2] + red[3];
}

// ---------------- vMF sampling (normalize + Wood accept/reject + Householder) ----------------
__global__ __launch_bounds__(128)
void vmf_sample_kernel(const float* __restrict__ z1, const float* __restrict__ z2,
                       __bf16* __restrict__ s1, __bf16* __restrict__ s2)
{
    __shared__ float red[4];
    __shared__ float wsh;
    const int d   = threadIdx.x;
    const int b   = blockIdx.x;
    const int s   = blockIdx.y;
    const int src = blockIdx.z;                      // 0 -> z1/s1, 1 -> z2/s2

    const float* z  = (src == 0 ? z1 : z2) + (size_t)b * DD;
    __bf16*     out = (src == 0 ? s1 : s2) + ((size_t)s * BB + (size_t)b) * DD;

    float zd  = z[d];
    float n2  = blockReduceSum128(zd * zd, red);
    float mu  = zd * __frsqrt_rn(n2);

    unsigned base = 0x2545F491u ^ (unsigned)src * 0x9e3779b9u
                                ^ (unsigned)s   * 0x85ebca6bu
                                ^ (unsigned)b   * 0xc2b2ae35u;

    if (d == 0) {   // scalar w ~ Wood's scheme, 8 proposal rounds, first accept
        const float m1  = 127.0f;
        const float kap = 32.0f;
        const float sq  = sqrtf(4.0f * kap * kap + m1 * m1);
        const float bw  = (-2.0f * kap + sq) / m1;
        const float aw  = (m1 + 2.0f * kap + sq) * 0.25f;
        const float dw  = 4.0f * aw * bw / (1.0f + bw) - m1 * __logf(m1);
        const float sigma = __frsqrt_rn(4.0f * (m1 + 1.0f));
        float w = 0.0f; bool done = false;
        #pragma unroll
        for (int r = 0; r < 8; ++r) {
            unsigned h1 = hashu(base ^ (0xA511E9B3u + (unsigned)r * 0x632BE5ABu));
            unsigned h2 = hashu(h1 ^ 0x68bc21ebu);
            unsigned h3 = hashu(h2 ^ 0x02e5be93u);
            float nrm = sqrtf(-2.0f * __logf(u01(h1))) * __cosf(6.28318530718f * u01(h2));
            float eps = fminf(fmaxf(0.5f + sigma * nrm, 1e-6f), 1.0f - 1e-6f);
            float den = 1.0f - (1.0f - bw) * eps;
            float wp  = (1.0f - (1.0f + bw) * eps) / den;
            float t   = 2.0f * aw * bw / den;
            bool acc  = (m1 * __logf(t) - t + dw) >= __logf(u01(h3));
            if (!done && acc) { w = wp; done = true; }
            if (r == 7 && !done) w = wp;
        }
        wsh = w;
    }
    __syncthreads();
    const float w = wsh;

    unsigned hg1 = hashu(base ^ (0x7f4a7c15u + (unsigned)d * 0x9E3779B9u));
    unsigned hg2 = hashu(hg1 + 0x165667b1u);
    float g = sqrtf(-2.0f * __logf(u01(hg1))) * __cosf(6.28318530718f * u01(hg2));
    if (d == 0) g = 0.0f;
    float gn2  = blockReduceSum128(g * g, red);
    float vfac = sqrtf(fmaxf(1.0f - w * w, 0.0f)) * __frsqrt_rn(gn2 + 1e-30f);
    float zz   = (d == 0) ? w : vfac * g;

    float uhd = ((d == 0) ? 1.0f : 0.0f) - mu;
    float un2 = blockReduceSum128(uhd * uhd, red);
    uhd *= __frsqrt_rn(un2 + 1e-24f);
    float dot = blockReduceSum128(zz * uhd, red);
    float o   = zz - 2.0f * dot * uhd;
    out[d] = (__bf16)o;
}

// ---------------- fused scores + row logsumexp ----------------
// Block: 128 thr / 4 waves; each wave owns TWO resident 16-row A tiles (128 rows/block).
// s2 panels (64 cols x 128 K, 16 KB) are double-buffered in LDS via CDNA5
// global_load_async_to_lds_b128 (ASYNCcnt pipeline) so the DMA of panel n+1
// overlaps all WMMA + exp work on panel n. Each LDS B fragment feeds 2 WMMAs (one per A tile).
__global__ __launch_bounds__(128)
void mcinfo_scores_lse_kernel(const __bf16* __restrict__ s1,
                              const __bf16* __restrict__ s2,
                              float* __restrict__ ratio)
{
    __shared__ __bf16 panel[2][PANEL_ELEMS];        // 2 x 16 KB

    const int lane  = threadIdx.x & 31;
    const int wave  = threadIdx.x >> 5;
    const int s     = blockIdx.y;
    const int rbase = blockIdx.x * 128 + wave * 32;   // this wave's 32 rows

    const __bf16* As = s1 + (size_t)s * BB * DD;
    const __bf16* Bs = s2 + (size_t)s * BB * DD;

    // Resident A fragments: 2 tiles x 4 k-steps (ISA 16-bit A layout)
    const int kA = (lane < 16) ? 0 : 8;
    v16bf a[2][4];
    #pragma unroll
    for (int t = 0; t < 2; ++t) {
        const int mrow = rbase + t * 16 + (lane & 15);
        #pragma unroll
        for (int kk = 0; kk < 4; ++kk) {
            const __bf16* p = As + (size_t)mrow * DD + kk * 32 + kA;
            ((v8bf*)&a[t][kk])[0] = *(const v8bf*)(p);
            ((v8bf*)&a[t][kk])[1] = *(const v8bf*)(p + 16);
        }
    }

    const int kBo  = (lane < 16) ? 0 : 8;
    const int ncol = lane & 15;

    float sums[2][8];
    #pragma unroll
    for (int t = 0; t < 2; ++t)
        #pragma unroll
        for (int r = 0; r < 8; ++r) sums[t][r] = 0.0f;
    float pos[2] = {0.0f, 0.0f};

    // async panel issue: 1024 b128 chunks / 128 threads = 8 per thread
    const unsigned lds0 = (unsigned)(uintptr_t)(&panel[0][0]);
    const unsigned lds1 = (unsigned)(uintptr_t)(&panel[1][0]);
    auto issue_panel = [&](int jp, int buf) {
        const char* g = (const char*)(Bs + (size_t)jp * JT * DD);
        const unsigned lb = buf ? lds1 : lds0;
        #pragma unroll
        for (int k = 0; k < 8; ++k) {
            const unsigned off = ((unsigned)threadIdx.x + k * 128u) * 16u;
            async_g2l_b128(lb + off, g + off);
        }
    };

    issue_panel(0, 0);

    const int NP = BB / JT;                           // 64 panels
    for (int jp = 0; jp < NP; ++jp) {
        const int buf = jp & 1;
        if (jp + 1 < NP) {
            issue_panel(jp + 1, buf ^ 1);
            S_WAIT_ASYNCCNT(8);                       // retire panel jp (in-order)
        } else {
            S_WAIT_ASYNCCNT(0);
        }
        __syncthreads();                              // panel jp visible to all waves

        const __bf16* pb = &panel[buf][0];
        #pragma unroll
        for (int sub = 0; sub < 4; ++sub) {
            const int j0g = jp * JT + sub * 16;
            v16bf bf[4];
            #pragma unroll
            for (int kk = 0; kk < 4; ++kk) {
                const __bf16* p = pb + (size_t)(sub * 16 + ncol) * DD + kk * 32 + kBo;
                ((v8bf*)&bf[kk])[0] = *(const v8bf*)(p);
                ((v8bf*)&bf[kk])[1] = *(const v8bf*)(p + 16);
            }
            #pragma unroll
            for (int t = 0; t < 2; ++t) {
                v8f c = {0.f, 0.f, 0.f, 0.f, 0.f, 0.f, 0.f, 0.f};
                #pragma unroll
                for (int kk = 0; kk < 4; ++kk)
                    c = __builtin_amdgcn_wmma_f32_16x16x32_bf16(false, a[t][kk], false, bf[kk],
                                                                (short)0, c, false, false);
                float tv[8];
                #pragma unroll
                for (int r = 0; r < 8; ++r) { tv[r] = 20.0f * c[r]; sums[t][r] += __expf(tv[r]); }

                if (j0g == rbase + t * 16) {          // diagonal tile
                    #pragma unroll
                    for (int r = 0; r < 8; ++r) {
                        pos[t] = (lane == r)      ? tv[r] : pos[t];
                        pos[t] = (lane == 24 + r) ? tv[r] : pos[t];
                    }
                }
            }
        }
        __syncthreads();                              // done reading panel[buf]
    }

    // combine row partial sums across the 16 lanes sharing each row group
    const float lB = __logf((float)BB);
    #pragma unroll
    for (int t = 0; t < 2; ++t) {
        #pragma unroll
        for (int r = 0; r < 8; ++r) {
            #pragma unroll
            for (int m = 8; m >= 1; m >>= 1) sums[t][r] += __shfl_xor(sums[t][r], m, 32);
        }
        const int i0t = rbase + t * 16;
        if (lane < 8) {
            float sr = sums[t][0];
            #pragma unroll
            for (int r = 1; r < 8; ++r) sr = (lane == r) ? sums[t][r] : sr;
            ratio[(size_t)s * BB + i0t + lane] = pos[t] - __logf(sr) + lB;
        } else if (lane >= 24) {
            const int r0 = lane - 24;
            float sr = sums[t][0];
            #pragma unroll
            for (int r = 1; r < 8; ++r) sr = (r0 == r) ? sums[t][r] : sr;
            ratio[(size_t)s * BB + i0t + 8 + r0] = pos[t] - __logf(sr) + lB;
        }
    }
}

// ---------------- final reduction ----------------
__global__ void zero_out_kernel(float* out) { out[0] = 0.0f; }

__global__ __launch_bounds__(256)
void finalize_kernel(const float* __restrict__ ratio, float* __restrict__ out)
{
    __shared__ float red[8];
    const int i = blockIdx.x * 256 + threadIdx.x;
    float acc = 0.0f;
    #pragma unroll
    for (int s = 0; s < SS; ++s) acc += __expf(ratio[(size_t)s * BB + i]);
    float v = -(__logf(acc) - __logf((float)SS));
    #pragma unroll
    for (int m = 16; m >= 1; m >>= 1) v += __shfl_xor(v, m, 32);
    if ((threadIdx.x & 31) == 0) red[threadIdx.x >> 5] = v;
    __syncthreads();
    if (threadIdx.x == 0) {
        float t = 0.0f;
        #pragma unroll
        for (int k = 0; k < 8; ++k) t += red[k];
        atomicAdd(out, t * (1.0f / (float)BB));
    }
}

extern "C" void kernel_launch(void* const* d_in, const int* in_sizes, int n_in,
                              void* d_out, int out_size, void* d_ws, size_t ws_size,
                              hipStream_t stream)
{
    const float* z1 = (const float*)d_in[0];
    const float* z2 = (const float*)d_in[1];
    float* out = (float*)d_out;

    char* ws = (char*)d_ws;
    const size_t sampBytes = (size_t)SS * BB * DD * sizeof(__bf16);   // 8 MB each
    __bf16* s1    = (__bf16*)(ws);
    __bf16* s2    = (__bf16*)(ws + sampBytes);
    float*  ratio = (float*)(ws + 2 * sampBytes);                     // 128 KB

    dim3 gs(BB, SS, 2);
    vmf_sample_kernel<<<gs, 128, 0, stream>>>(z1, z2, s1, s2);

    dim3 gw(BB / 128, SS, 1);
    mcinfo_scores_lse_kernel<<<gw, 128, 0, stream>>>(s1, s2, ratio);

    zero_out_kernel<<<1, 1, 0, stream>>>(out);
    finalize_kernel<<<BB / 256, 256, 0, stream>>>(ratio, out);
}
